// LoRAFlashGQA_10883447128123
// MI455X (gfx1250) — matrix operand order repaired and checked by reading, not compiled, not run on hardware
//
#include <hip/hip_runtime.h>
#include <hip/hip_bf16.h>

#define D_MODEL 2048
#define N_HEADS 16
#define N_KV    4
#define D_K     128
#define KV_DIM  512
#define RANK    16
#define SCALING 2.0f
#define BSZ     2
#define SEQ     2048
#define MROWS   (BSZ*SEQ)   // 4096

typedef __attribute__((ext_vector_type(16))) __bf16 bf16x16;
typedef __attribute__((ext_vector_type(8)))  float  f32x8;
typedef unsigned int u32x4 __attribute__((ext_vector_type(4)));
typedef int i32x4 __attribute__((ext_vector_type(4)));
typedef int i32x8 __attribute__((ext_vector_type(8)));

union FragB16 { bf16x16 v; uint4 q[2]; };

#if defined(__HIP_DEVICE_COMPILE__) && __has_builtin(__builtin_amdgcn_tensor_load_to_lds) && __has_builtin(__builtin_amdgcn_s_wait_tensorcnt)
#define USE_TDM 1
#endif

#ifdef USE_TDM
// ---------------------------------------------------------------------------
// Tensor Data Mover: 2D tile global -> LDS with per-row LDS padding.
//   elems   : 8-byte elements per row (tile_dim0 / tensor_dim0)
//   rows    : rows (tile_dim1 / tensor_dim1)
//   stride8 : global row stride in 8-byte units (tensor_dim0_stride)
//   pad_int : encoded pad interval (DWORDs before pad = 2^(pad_int+1))
//   pad_amt : encoded pad amount   (pad DWORDs = pad_amt+1)
// D# layout per CDNA5 ISA ch.8 (group0/1; groups 2/3 zero for 2D tensors).
// ---------------------------------------------------------------------------
__device__ inline void tdm_load_2d(unsigned lds_off, const void* gaddr,
                                   unsigned elems, unsigned rows,
                                   unsigned stride8,
                                   unsigned pad_int, unsigned pad_amt) {
  unsigned long long ga = (unsigned long long)(uintptr_t)gaddr;
  u32x4 g0;
  g0[0] = 1u;                                    // count=1 (valid user D#)
  g0[1] = lds_off;                               // lds_addr (bytes)
  g0[2] = (unsigned)ga;                          // global_addr[31:0]
  g0[3] = (unsigned)(ga >> 32) | (2u << 30);     // global_addr[56:32] | type=2
  i32x8 g1;
  g1[0] = (int)((3u << 16)                       // data_size = 8B
              | (1u << 20)                       // pad_enable
              | (pad_int << 22) | (pad_amt << 25));
  g1[1] = (int)(elems << 16);                    // tensor_dim0[15:0] @ [63:48]
  g1[2] = (int)(rows << 16);                     // td0 hi=0, tensor_dim1[15:0]
  g1[3] = (int)(elems << 16);                    // td1 hi=0, tile_dim0
  g1[4] = (int)rows;                             // tile_dim1, tile_dim2=0
  g1[5] = (int)stride8;                          // tensor_dim0_stride[31:0]
  g1[6] = 0;                                     // stride hi, dim1_stride lo
  g1[7] = 0;
  i32x4 g2 = {0, 0, 0, 0}, g3 = {0, 0, 0, 0};
#if __clang_major__ >= 23
  i32x8 g4 = {0, 0, 0, 0, 0, 0, 0, 0};
  __builtin_amdgcn_tensor_load_to_lds(g0, g1, g2, g3, g4, 0);
#else
  __builtin_amdgcn_tensor_load_to_lds(g0, g1, g2, g3, 0);
#endif
}
__device__ inline unsigned lds_off_of(const void* p) {
  return (unsigned)(uintptr_t)p;   // generic LDS ptr low 32 bits = wave LDS addr
}
#endif

// ---------------------------------------------------------------------------
// Elementwise: f32 -> bf16 cast
// ---------------------------------------------------------------------------
__global__ void cast_f32_bf16(const float* __restrict__ src, __bf16* __restrict__ dst, int n) {
  int i = blockIdx.x * blockDim.x + threadIdx.x;
  if (i < n) dst[i] = (__bf16)src[i];
}

// ---------------------------------------------------------------------------
// Fuse LoRA into weight: W_eff[o][i] = W[o][i] + SCALING * sum_r B[r][o]*A[i][r]
// ---------------------------------------------------------------------------
__global__ void lora_weight(const float* __restrict__ W, const float* __restrict__ A,
                            const float* __restrict__ Bm, __bf16* __restrict__ out,
                            int outDim, int inDim) {
  int idx = blockIdx.x * blockDim.x + threadIdx.x;
  if (idx >= outDim * inDim) return;
  int o = idx / inDim, i = idx % inDim;
  float acc = W[idx];
#pragma unroll
  for (int r = 0; r < RANK; ++r)
    acc += SCALING * Bm[r * outDim + o] * A[i * RANK + r];
  out[idx] = (__bf16)acc;
}

// ---------------------------------------------------------------------------
// NT GEMM: C[M,N] = A[M,K] * B[N,K]^T, bf16 WMMA, f32 accum, double-buffered.
// Block 256 threads (8 waves), tile 128(M) x 64(N), K step 32.
// ---------------------------------------------------------------------------
template<bool OUT_F32>
__global__ __launch_bounds__(256)
void gemm_nt(const __bf16* __restrict__ A, const __bf16* __restrict__ B,
             void* __restrict__ Cv, int M, int N, int K) {
  __shared__ __align__(16) __bf16 As[2][128 * 40];   // 32 k + 8 pad
  __shared__ __align__(16) __bf16 Bs[2][64 * 40];
  const int tid  = threadIdx.x;
  const int lane = tid & 31, w = tid >> 5;
  const int lm   = lane & 15, g = lane >> 4;
  const int m0 = blockIdx.y * 128, n0 = blockIdx.x * 64;
  const int wm = (w & 3) * 32, wn = (w >> 2) * 32;

  auto stage = [&](int kk, int buf) {
#pragma unroll
    for (int i = 0; i < 2; ++i) {
      int c = tid + i * 256;
      int row = c >> 2, kc = c & 3;
      *(uint4*)&As[buf][row * 40 + kc * 8] =
          *(const uint4*)&A[(size_t)(m0 + row) * K + kk + kc * 8];
    }
    int row = tid >> 2, kc = tid & 3;
    *(uint4*)&Bs[buf][row * 40 + kc * 8] =
        *(const uint4*)&B[(size_t)(n0 + row) * K + kk + kc * 8];
  };

  f32x8 zero8 = {0.f,0.f,0.f,0.f,0.f,0.f,0.f,0.f};
  f32x8 acc[2][2];
  acc[0][0] = zero8; acc[0][1] = zero8; acc[1][0] = zero8; acc[1][1] = zero8;

  stage(0, 0);
  const int nk = K >> 5;
  for (int it = 0; it < nk; ++it) {
    __syncthreads();
    if (it + 1 < nk) stage((it + 1) * 32, (it + 1) & 1);   // overlap next stage
    const int buf = it & 1;

    FragB16 fa[2], fb[2];
#pragma unroll
    for (int mi = 0; mi < 2; ++mi) {       // A-frag: 16x32 lane-split layout
      const __bf16* p = &As[buf][(wm + mi * 16 + lm) * 40 + g * 8];
      fa[mi].q[0] = *(const uint4*)p;
      fa[mi].q[1] = *(const uint4*)(p + 16);
    }
#pragma unroll
    for (int ni = 0; ni < 2; ++ni) {       // B-frag: 16x32 lane-split layout
      const __bf16* p = &Bs[buf][(wn + ni * 16 + lm) * 40 + g * 8];
      fb[ni].q[0] = *(const uint4*)p;
      fb[ni].q[1] = *(const uint4*)(p + 16);
    }
#pragma unroll
    for (int mi = 0; mi < 2; ++mi)
#pragma unroll
      for (int ni = 0; ni < 2; ++ni)
        acc[mi][ni] = __builtin_amdgcn_wmma_f32_16x16x32_bf16(
            false, fa[mi].v, false, fb[ni].v, (short)0, acc[mi][ni], false, false);
  }

  float*  Cf = (float*)Cv;
  __bf16* Cb = (__bf16*)Cv;
#pragma unroll
  for (int mi = 0; mi < 2; ++mi)
#pragma unroll
    for (int ni = 0; ni < 2; ++ni)
#pragma unroll
      for (int r = 0; r < 8; ++r) {
        int row = m0 + wm + mi * 16 + r + 8 * g;
        int col = n0 + wn + ni * 16 + lm;
        float v = acc[mi][ni][r];
        if (OUT_F32) Cf[(size_t)row * N + col] = v;
        else         Cb[(size_t)row * N + col] = (__bf16)v;
      }
}

// ---------------------------------------------------------------------------
// RoPE + permute [b,s,h,d] -> [b,h,s,d].  128 threads = one (b,h,s) row.
// ---------------------------------------------------------------------------
__global__ void rope_perm(const __bf16* __restrict__ src, const float* __restrict__ cs,
                          const float* __restrict__ sn, __bf16* __restrict__ dst, int nh) {
  int d   = threadIdx.x;
  int blk = blockIdx.x;                  // (b*nh + h)*SEQ + s
  int s  = blk % SEQ;
  int bh = blk / SEQ;
  int h  = bh % nh, b = bh / nh;
  size_t si = (((size_t)b * SEQ + s) * nh + h) * D_K;
  float qv  = (float)src[si + d];
  int d2 = (d < 64) ? d + 64 : d - 64;
  float q2 = (float)src[si + d2];
  float rot = (d < 64) ? -q2 : q2;
  float o = qv * cs[s * D_K + d] + rot * sn[s * D_K + d];
  dst[(((size_t)b * nh + h) * SEQ + s) * D_K + d] = (__bf16)o;
}

// ---------------------------------------------------------------------------
// V transpose: [b,s,kh,d] -> [b,kh,d,s]
// ---------------------------------------------------------------------------
__global__ void v_transpose(const __bf16* __restrict__ src, __bf16* __restrict__ dst) {
  int d   = threadIdx.x;
  int blk = blockIdx.x;                  // (b*N_KV + kh)*SEQ + s
  int s  = blk % SEQ;
  int bk = blk / SEQ;
  int kh = bk % N_KV, b = bk / N_KV;
  __bf16 v = src[(((size_t)b * SEQ + s) * N_KV + kh) * D_K + d];
  dst[(((size_t)b * N_KV + kh) * D_K + d) * SEQ + s] = v;
}

// ---------------------------------------------------------------------------
// Causal GQA flash attention.
// Grid: (SEQ/64, N_HEADS, BSZ). Block: 128 threads (4 waves), 64-query tile.
// K/V tiles staged via Tensor Data Mover (fallback: cooperative b128 loads).
// ---------------------------------------------------------------------------
__global__ __launch_bounds__(128)
void flash_attn(const __bf16* __restrict__ Qh, const __bf16* __restrict__ Kh,
                const __bf16* __restrict__ Vt, __bf16* __restrict__ ctx) {
  __shared__ __align__(16) __bf16 Qs[64 * 136];   // [q][d] + 8 pad
  __shared__ __align__(16) __bf16 Ks[64 * 136];   // [key][d] + 8 pad
  __shared__ __align__(16) __bf16 Vs[128 * 72];   // [d][key] + 8 pad
  __shared__ __align__(16) __bf16 Ps[4][16 * 72]; // per-wave P strip

  const int tid  = threadIdx.x;
  const int lane = tid & 31, w = tid >> 5;
  const int lm   = lane & 15, g = lane >> 4;
  const int qt = blockIdx.x, h = blockIdx.y, b = blockIdx.z;
  const int kh = h >> 2;                 // group = N_HEADS / N_KV = 4
  const int qbase = qt * 64;

  const __bf16* Qg = Qh + (((size_t)b * N_HEADS + h) * SEQ + qbase) * D_K;
  const __bf16* Kg = Kh + ((size_t)b * N_KV + kh) * SEQ * D_K;
  const __bf16* Vg = Vt + ((size_t)b * N_KV + kh) * (size_t)D_K * SEQ;

  // ---- Stage Q tile (64 x 128) ----
#ifdef USE_TDM
  if (w == 0) {
    // 64 rows of 256B (32 x 8B), LDS rows padded to 272B (pad 4 DW per 64 DW)
    tdm_load_2d(lds_off_of(Qs), Qg, 32, 64, 32, 5, 3);
    __builtin_amdgcn_s_wait_tensorcnt(0);
  }
#else
#pragma unroll
  for (int i = 0; i < 8; ++i) {
    int c = tid + i * 128;
    int row = c >> 4, dc = c & 15;
    *(uint4*)&Qs[row * 136 + dc * 8] = *(const uint4*)&Qg[(size_t)row * D_K + dc * 8];
  }
#endif
  __syncthreads();

  FragB16 qa[4];
#pragma unroll
  for (int kf = 0; kf < 4; ++kf) {
    const __bf16* p = &Qs[(w * 16 + lm) * 136 + kf * 32 + g * 8];
    qa[kf].q[0] = *(const uint4*)p;
    qa[kf].q[1] = *(const uint4*)(p + 16);
  }

  f32x8 zero8 = {0.f,0.f,0.f,0.f,0.f,0.f,0.f,0.f};
  f32x8 accO[8];
#pragma unroll
  for (int t = 0; t < 8; ++t) accO[t] = zero8;
  float mrow[8], lrow[8];
#pragma unroll
  for (int r = 0; r < 8; ++r) { mrow[r] = -1e30f; lrow[r] = 0.f; }

  const float scale = 0.08838834764831845f;  // 1/sqrt(128)
  const int ntiles = qt + 1;                 // causal

  for (int jt = 0; jt < ntiles; ++jt) {
    __syncthreads();
    // ---- Stage K tile (64 keys x 128 d) and V tile (128 d x 64 keys) ----
#ifdef USE_TDM
    if (w == 0) {
      tdm_load_2d(lds_off_of(Ks), Kg + (size_t)(jt * 64) * D_K, 32, 64, 32, 5, 3);
      // V rows: 128B (16 x 8B), global stride SEQ*2B = 512 x 8B,
      // LDS rows padded to 144B (pad 4 DW per 32 DW)
      tdm_load_2d(lds_off_of(Vs), Vg + jt * 64, 16, 128, 512, 4, 3);
      __builtin_amdgcn_s_wait_tensorcnt(0);
    }
#else
#pragma unroll
    for (int i = 0; i < 8; ++i) {
      int c = tid + i * 128;
      int row = c >> 4, dc = c & 15;
      *(uint4*)&Ks[row * 136 + dc * 8] =
          *(const uint4*)&Kg[((size_t)(jt * 64 + row)) * D_K + dc * 8];
    }
#pragma unroll
    for (int i = 0; i < 8; ++i) {
      int c = tid + i * 128;
      int row = c >> 3, sc = c & 7;
      *(uint4*)&Vs[row * 72 + sc * 8] =
          *(const uint4*)&Vg[(size_t)row * SEQ + jt * 64 + sc * 8];
    }
#endif
    __syncthreads();

    // ---- S = Q K^T (16 q x 64 keys per wave) ----
    f32x8 sacc[4];
#pragma unroll
    for (int nt = 0; nt < 4; ++nt) sacc[nt] = zero8;
#pragma unroll
    for (int nt = 0; nt < 4; ++nt)
#pragma unroll
      for (int kf = 0; kf < 4; ++kf) {
        FragB16 kb;
        const __bf16* p = &Ks[(nt * 16 + lm) * 136 + kf * 32 + g * 8];
        kb.q[0] = *(const uint4*)p;
        kb.q[1] = *(const uint4*)(p + 16);
        sacc[nt] = __builtin_amdgcn_wmma_f32_16x16x32_bf16(
            false, qa[kf].v, false, kb.v, (short)0, sacc[nt], false, false);
      }

    // ---- Scale + causal mask (diagonal tile only) ----
    const bool diag = (jt == qt);
#pragma unroll
    for (int nt = 0; nt < 4; ++nt)
#pragma unroll
      for (int r = 0; r < 8; ++r) {
        float v = sacc[nt][r] * scale;
        if (diag) {
          int col = nt * 16 + lm;
          int row = w * 16 + r + 8 * g;
          if (col > row) v = -1e30f;
        }
        sacc[nt][r] = v;
      }

    // ---- Online softmax (row lives in one 16-lane half) ----
    float mnew[8], csc[8], rs[8];
#pragma unroll
    for (int r = 0; r < 8; ++r) {
      float v = fmaxf(fmaxf(sacc[0][r], sacc[1][r]), fmaxf(sacc[2][r], sacc[3][r]));
      v = fmaxf(v, __shfl_xor(v, 1, 32));
      v = fmaxf(v, __shfl_xor(v, 2, 32));
      v = fmaxf(v, __shfl_xor(v, 4, 32));
      v = fmaxf(v, __shfl_xor(v, 8, 32));
      mnew[r] = fmaxf(mrow[r], v);
      csc[r]  = __expf(mrow[r] - mnew[r]);
      mrow[r] = mnew[r];
      rs[r]   = 0.f;
    }
#pragma unroll
    for (int nt = 0; nt < 4; ++nt)
#pragma unroll
      for (int r = 0; r < 8; ++r) {
        float p = __expf(sacc[nt][r] - mnew[r]);
        sacc[nt][r] = p;
        rs[r] += p;
      }
#pragma unroll
    for (int r = 0; r < 8; ++r) {
      float v = rs[r];
      v += __shfl_xor(v, 1, 32);
      v += __shfl_xor(v, 2, 32);
      v += __shfl_xor(v, 4, 32);
      v += __shfl_xor(v, 8, 32);
      lrow[r] = lrow[r] * csc[r] + v;
    }
#pragma unroll
    for (int t = 0; t < 8; ++t)
#pragma unroll
      for (int r = 0; r < 8; ++r)
        accO[t][r] *= csc[r];

    // ---- P (C layout) -> LDS -> A-fragment layout ----
#pragma unroll
    for (int nt = 0; nt < 4; ++nt)
#pragma unroll
      for (int r = 0; r < 8; ++r)
        Ps[w][(r + 8 * g) * 72 + nt * 16 + lm] = (__bf16)sacc[nt][r];
    __syncthreads();

    FragB16 pa[2];
#pragma unroll
    for (int kf = 0; kf < 2; ++kf) {
      const __bf16* p = &Ps[w][lm * 72 + kf * 32 + g * 8];
      pa[kf].q[0] = *(const uint4*)p;
      pa[kf].q[1] = *(const uint4*)(p + 16);
    }
    // ---- O += P V (16 q x 128 d) ----
#pragma unroll
    for (int t = 0; t < 8; ++t)
#pragma unroll
      for (int kf = 0; kf < 2; ++kf) {
        FragB16 vb;
        const __bf16* p = &Vs[(t * 16 + lm) * 72 + kf * 32 + g * 8];
        vb.q[0] = *(const uint4*)p;
        vb.q[1] = *(const uint4*)(p + 16);
        accO[t] = __builtin_amdgcn_wmma_f32_16x16x32_bf16(
            false, pa[kf].v, false, vb.v, (short)0, accO[t], false, false);
      }
  }

  // ---- Epilogue: O / l -> ctx [b, s, h*128 + d] ----
#pragma unroll
  for (int r = 0; r < 8; ++r) {
    float inv = 1.0f / lrow[r];
    int srow = qbase + w * 16 + r + 8 * g;
    size_t base = ((size_t)b * SEQ + srow) * D_MODEL + h * D_K;
#pragma unroll
    for (int t = 0; t < 8; ++t)
      ctx[base + t * 16 + lm] = (__bf16)(accO[t][r] * inv);
  }
}

// ---------------------------------------------------------------------------
extern "C" void kernel_launch(void* const* d_in, const int* in_sizes, int n_in,
                              void* d_out, int out_size, void* d_ws, size_t ws_size,
                              hipStream_t stream) {
  const float* x   = (const float*)d_in[0];
  const float* fc  = (const float*)d_in[1];
  const float* fs  = (const float*)d_in[2];
  const float* W_q = (const float*)d_in[3];
  const float* W_k = (const float*)d_in[4];
  const float* W_v = (const float*)d_in[5];
  const float* W_o = (const float*)d_in[6];
  const float* A_q = (const float*)d_in[7];
  const float* B_q = (const float*)d_in[8];
  const float* A_k = (const float*)d_in[9];
  const float* B_k = (const float*)d_in[10];
  const float* A_v = (const float*)d_in[11];
  const float* B_v = (const float*)d_in[12];

  char* ws = (char*)d_ws;
  size_t off = 0;
  auto alloc = [&](size_t bytes) {
    void* p = ws + off;
    off = (off + bytes + 255) & ~(size_t)255;
    return p;
  };
  __bf16* xb  = (__bf16*)alloc((size_t)MROWS  * D_MODEL * 2);
  __bf16* Wqb = (__bf16*)alloc((size_t)D_MODEL * D_MODEL * 2);
  __bf16* Wkb = (__bf16*)alloc((size_t)KV_DIM  * D_MODEL * 2);
  __bf16* Wvb = (__bf16*)alloc((size_t)KV_DIM  * D_MODEL * 2);
  __bf16* Wob = (__bf16*)alloc((size_t)D_MODEL * D_MODEL * 2);
  __bf16* Qt  = (__bf16*)alloc((size_t)MROWS  * D_MODEL * 2);
  __bf16* Kt  = (__bf16*)alloc((size_t)MROWS  * KV_DIM  * 2);
  __bf16* Vtm = (__bf16*)alloc((size_t)MROWS  * KV_DIM  * 2);
  __bf16* Qh  = (__bf16*)alloc((size_t)MROWS  * D_MODEL * 2);
  __bf16* Kh  = (__bf16*)alloc((size_t)MROWS  * KV_DIM  * 2);
  __bf16* Vh  = (__bf16*)alloc((size_t)MROWS  * KV_DIM  * 2);
  __bf16* ctx = Qt;   // Qt dead after rope_perm -> reuse as ctx

  // 1. casts + LoRA weight fusion
  {
    int n = MROWS * D_MODEL;
    cast_f32_bf16<<<n / 256, 256, 0, stream>>>(x, xb, n);
  }
  lora_weight<<<(D_MODEL * D_MODEL) / 256, 256, 0, stream>>>(W_q, A_q, B_q, Wqb, D_MODEL, D_MODEL);
  lora_weight<<<(KV_DIM * D_MODEL) / 256, 256, 0, stream>>>(W_k, A_k, B_k, Wkb, KV_DIM, D_MODEL);
  lora_weight<<<(KV_DIM * D_MODEL) / 256, 256, 0, stream>>>(W_v, A_v, B_v, Wvb, KV_DIM, D_MODEL);
  cast_f32_bf16<<<(D_MODEL * D_MODEL) / 256, 256, 0, stream>>>(W_o, Wob, D_MODEL * D_MODEL);

  // 2. QKV projections (bf16 WMMA)
  gemm_nt<false><<<dim3(D_MODEL / 64, MROWS / 128), 256, 0, stream>>>(xb, Wqb, Qt, MROWS, D_MODEL, D_MODEL);
  gemm_nt<false><<<dim3(KV_DIM  / 64, MROWS / 128), 256, 0, stream>>>(xb, Wkb, Kt, MROWS, KV_DIM,  D_MODEL);
  gemm_nt<false><<<dim3(KV_DIM  / 64, MROWS / 128), 256, 0, stream>>>(xb, Wvb, Vtm, MROWS, KV_DIM, D_MODEL);

  // 3. RoPE + head-major permutes; V -> [b,kh,d,s]
  rope_perm<<<BSZ * N_HEADS * SEQ, 128, 0, stream>>>(Qt, fc, fs, Qh, N_HEADS);
  rope_perm<<<BSZ * N_KV   * SEQ, 128, 0, stream>>>(Kt, fc, fs, Kh, N_KV);
  v_transpose<<<BSZ * N_KV * SEQ, 128, 0, stream>>>(Vtm, Vh);

  // 4. Causal GQA flash attention -> ctx [b,s,D_MODEL] bf16
  flash_attn<<<dim3(SEQ / 64, N_HEADS, BSZ), 128, 0, stream>>>(Qh, Kh, Vh, ctx);

  // 5. Output projection -> f32 d_out
  gemm_nt<true><<<dim3(D_MODEL / 64, MROWS / 128), 256, 0, stream>>>(ctx, Wob, d_out, MROWS, D_MODEL, D_MODEL);
}
